// ModConv3d_43885975831075
// MI455X (gfx1250) — matrix operand order repaired and checked by reading, not compiled, not run on hardware
//
#include <hip/hip_runtime.h>

// ---------------------------------------------------------------------------
// Modulated 3D conv (StyleGAN-style) as implicit GEMM on CDNA5 WMMA.
//   M = c_out = 128, N = spatial = 32768/sample, K = c_in*27 = 3456
//   bf16 A/B, f32 accumulate via v_wmma_f32_16x16x32_bf16 (wave32).
// Block tile: 128 (M) x 64 (N: two W-rows). 8 waves; wave w owns M-rows
// w*16..w*16+15 and all four 16-wide N tiles (A fragment reused 4x).
// Double-buffered LDS im2col tile; unconditional pipelined global loads so
// the WMMAs only need loadcnt<=8 (A done, prefetch still in flight); all
// B fragments loaded before the WMMA block (sched_barrier) so the four
// WMMAs issue back-to-back after a single dscnt wait.
// ---------------------------------------------------------------------------

typedef __attribute__((ext_vector_type(16))) __bf16 v16bf;
typedef __attribute__((ext_vector_type(8)))  float  v8f;

#define B_SZ     8
#define CIN      128
#define COUT     128
#define STYLE_D  512
#define TAPS     27          // 3*3*3
#define SPATIAL  32768       // 32^3
#define NTILE    64          // N per block (two y-rows of 32)
#define BS_ROW_U 20          // Bs row stride in uints  (40 bf16 = 80 B, 16B aligned, padded)
#define BS_ROW_Q 5           // Bs row stride in uint4
#define NCHUNK   108         // 27 taps * 4 channel chunks of 32

// exact RNE (used once, in the weight-prep kernel)
static __device__ __forceinline__ unsigned short f2bf(float f) {
  union { float f; unsigned u; } v; v.f = f;
  unsigned r = v.u + 0x7FFFu + ((v.u >> 16) & 1u);
  return (unsigned short)(r >> 16);
}

// fast path: round-to-nearest (+0x8000) then pack two high halves in one
// v_perm_b32. D.lo16 = bf16(a), D.hi16 = bf16(b).
static __device__ __forceinline__ unsigned pack_bf16(float a, float b) {
  union { float f; unsigned u; } ua, ub;
  ua.f = a; ub.f = b;
  unsigned ra = ua.u + 0x8000u;
  unsigned rb = ub.u + 0x8000u;
  return __builtin_amdgcn_perm(rb, ra, 0x07060302u);
}

// ---------------------------------------------------------------------------
// Stage 1: style[b][i] = s[b] . affine_w[i] + affine_b[i]
// ---------------------------------------------------------------------------
__global__ __launch_bounds__(128) void style_kernel(
    const float* __restrict__ s, const float* __restrict__ aw,
    const float* __restrict__ ab, float* __restrict__ style) {
  int b = blockIdx.x;
  int i = threadIdx.x;
  const float* sr = s  + b * STYLE_D;
  const float* wr = aw + i * STYLE_D;
  float acc = ab[i];
  #pragma unroll 4
  for (int d = 0; d < STYLE_D; ++d) acc += sr[d] * wr[d];
  style[b * CIN + i] = acc;
}

// ---------------------------------------------------------------------------
// Stage 2: modulate + demodulate, convert to bf16, relayout to [b][tap][o][i]
// ---------------------------------------------------------------------------
__global__ __launch_bounds__(128) void modw_kernel(
    const float* __restrict__ w, const float* __restrict__ style,
    unsigned short* __restrict__ wmod) {
  __shared__ float red[128];
  int b = blockIdx.x >> 7;
  int o = blockIdx.x & 127;
  int tid = threadIdx.x;

  float part = 0.0f;
  for (int e = tid; e < CIN * TAPS; e += 128) {
    int i = e / TAPS, tap = e - i * TAPS;
    float m  = style[b * CIN + i] + 1.0f;
    float wv = w[(o * CIN + i) * TAPS + tap] * m;
    part += wv * wv;
  }
  red[tid] = part;
  __syncthreads();
  for (int st = 64; st > 0; st >>= 1) {
    if (tid < st) red[tid] += red[tid + st];
    __syncthreads();
  }
  float scale = 1.0f / sqrtf(red[0] + 1e-8f);

  for (int e = tid; e < CIN * TAPS; e += 128) {
    int i = e / TAPS, tap = e - i * TAPS;
    float m  = style[b * CIN + i] + 1.0f;
    float wv = w[(o * CIN + i) * TAPS + tap] * m * scale;
    wmod[((size_t)((b * TAPS + tap) * COUT + o)) * CIN + i] = f2bf(wv);
  }
}

// ---------------------------------------------------------------------------
// Stage 3: implicit-GEMM conv, double-buffered + pipelined.
// ---------------------------------------------------------------------------
__global__ __launch_bounds__(256) void conv_kernel(
    const float* __restrict__ x, const unsigned short* __restrict__ wmod,
    const float* __restrict__ bias, float* __restrict__ out) {
  __shared__ uint4 BsQ[2][NTILE * BS_ROW_Q];   // 10 KB

  int b  = blockIdx.x >> 9;          // 512 blocks per sample
  int sp = blockIdx.x & 511;
  int z  = sp >> 4;                  // 0..31
  int y0 = (sp & 15) << 1;           // block covers rows y0, y0+1

  int lane = threadIdx.x & 31;
  int wv   = threadIdx.x >> 5;       // wave id == M-tile id
  int h    = lane >> 4;              // K-half selector
  int nlo  = lane & 15;

  // fill mapping: lanes sweep x for a fixed channel (coalesced 128B rows)
  int fn  = threadIdx.x & 63;
  int fc0 = (threadIdx.x >> 6) << 3;
  int fy  = y0 + (fn >> 5);
  int fx  = fn & 31;

  v8f acc[4] = {};

  const uint4* wq = (const uint4*)wmod;      // A row = 16 uint4 (128 bf16)
  int m    = wv * 16 + nlo;
  int aoff = ((b * TAPS) * COUT + m) << 4;

  // --- unconditional prefetch: safe base when OOB, values zeroed by select
  float pv[8];
  bool  pok;
  auto prefetch = [&](int dzi, int dyi, int kwi, int cc) {
    int zz = z  + dzi - 1;
    int yy = fy + dyi - 1;
    int xx = fx + kwi - 1;
    pok = ((unsigned)zz < 32u) & ((unsigned)yy < 32u) & ((unsigned)xx < 32u);
    size_t base = pok ? (((size_t)(b * CIN + cc * 32 + fc0) << 15)
                         + (zz << 10) + (yy << 5) + xx)
                      : (size_t)0;
    #pragma unroll
    for (int j = 0; j < 8; ++j) pv[j] = x[base + ((size_t)j << 15)];
  };

  prefetch(0, 0, 0, 0);
  bool ok0 = pok;

  int kwi = 0, dyi = 0, dzi = 0;     // incremental tap decode (no div/mod)
  for (int chunk = 0; chunk < NCHUNK; ++chunk) {
    int cc = chunk & 3;

    // ---- pack + store prefetched values into current buffer
    {
      unsigned* bsu = (unsigned*)BsQ[chunk & 1];
      int ub = fn * BS_ROW_U + (fc0 >> 1);
      #pragma unroll
      for (int j = 0; j < 4; ++j) {
        float a = ok0 ? pv[2 * j]     : 0.0f;
        float c = ok0 ? pv[2 * j + 1] : 0.0f;
        bsu[ub + j] = pack_bf16(a, c);
      }
    }
    __syncthreads();   // single barrier per chunk (double buffering)

    // ---- A fragment FIRST (so WMMA only needs loadcnt <= 8 later)
    union { v16bf v; uint4 q[2]; } afr;
    afr.q[0] = wq[aoff + cc * 4 + h];        // K = 8h + 0..7
    afr.q[1] = wq[aoff + cc * 4 + h + 2];    // K = 8h + 16..23

    // ---- issue next chunk's 8 global loads (in flight across the WMMAs)
    if (chunk + 1 < NCHUNK) {
      int nk = kwi, ndy = dyi, ndz = dzi;
      if (cc == 3) { if (++nk == 3) { nk = 0; if (++ndy == 3) { ndy = 0; ++ndz; } } }
      prefetch(ndz, ndy, nk, (chunk + 1) & 3);
    }
    bool okn = pok;

    // ---- all 4 B fragments resident before the WMMA block
    const uint4* __restrict__ bq = BsQ[chunk & 1];
    union { v16bf v; uint4 q[2]; } bfr[4];
    #pragma unroll
    for (int nt = 0; nt < 4; ++nt) {
      int r0 = (nt * 16 + nlo) * BS_ROW_Q + h;
      bfr[nt].q[0] = bq[r0];
      bfr[nt].q[1] = bq[r0 + 2];
    }
    // forbid the scheduler from sinking the ds_loads between the WMMAs:
    // one dscnt wait, then 4 back-to-back WMMAs on the XDL pipe.
    __builtin_amdgcn_sched_barrier(0);
    #pragma unroll
    for (int nt = 0; nt < 4; ++nt) {
      acc[nt] = __builtin_amdgcn_wmma_f32_16x16x32_bf16(
          false, afr.v, false, bfr[nt].v, (short)0, acc[nt], false, false);
    }
    ok0 = okn;

    if (cc == 3) {                   // advance tap
      aoff += COUT << 4;
      if (++kwi == 3) { kwi = 0; if (++dyi == 3) { dyi = 0; ++dzi; } }
    }
  }

  // ---- epilogue: C/D layout -> lane n = nlo (+16*nt), M = reg + 8h
  #pragma unroll
  for (int r = 0; r < 8; ++r) {
    int o = wv * 16 + h * 8 + r;
    float bv = bias[o];
    size_t base = ((size_t)(b * COUT + o) << 15) + (z << 10);
    #pragma unroll
    for (int nt = 0; nt < 4; ++nt) {
      int n = nt * 16 + nlo;
      out[base + ((y0 + (n >> 5)) << 5) + (n & 31)] = acc[nt][r] + bv;
    }
  }
}

// ---------------------------------------------------------------------------
extern "C" void kernel_launch(void* const* d_in, const int* in_sizes, int n_in,
                              void* d_out, int out_size, void* d_ws, size_t ws_size,
                              hipStream_t stream) {
  const float* x    = (const float*)d_in[0];   // [8,128,32,32,32]
  const float* s    = (const float*)d_in[1];   // [8,512]
  const float* w    = (const float*)d_in[2];   // [128,128,3,3,3]
  const float* bias = (const float*)d_in[3];   // [128]
  const float* aw   = (const float*)d_in[4];   // [128,512]
  const float* ab   = (const float*)d_in[5];   // [128]
  float* out = (float*)d_out;

  float*          style = (float*)d_ws;
  unsigned short* wmod  = (unsigned short*)((char*)d_ws + 4096);

  style_kernel<<<B_SZ, 128, 0, stream>>>(s, aw, ab, style);
  modw_kernel<<<B_SZ * COUT, 128, 0, stream>>>(w, style, wmod);
  conv_kernel<<<B_SZ * (SPATIAL / NTILE), 256, 0, stream>>>(x, wmod, bias, out);
}
